// SparseAutoencoder_77103252898272
// MI455X (gfx1250) — matrix operand order repaired and checked by reading, not compiled, not run on hardware
//
#include <hip/hip_runtime.h>
#include <hip/hip_bf16.h>

// ---------------------------------------------------------------------------
// Types for CDNA5 WMMA
// ---------------------------------------------------------------------------
typedef __attribute__((ext_vector_type(16))) __bf16 v16bf;
typedef __attribute__((ext_vector_type(8)))  float  v8f;

union AFrag {
    v16bf v;
    uint4 q[2];
};

// Problem constants
#define D_MODEL   768
#define N_LATENTS 16384
#define N_ROWS    8192
#define TOPK      32

// ---------------------------------------------------------------------------
// 1) float32 -> bf16 (round to nearest even) elementwise
// ---------------------------------------------------------------------------
__global__ void sae_f32_to_bf16(const float* __restrict__ in,
                                unsigned short* __restrict__ out, int n) {
    int i = blockIdx.x * blockDim.x + threadIdx.x;
    if (i < n) {
        unsigned int u = __float_as_uint(in[i]);
        unsigned int r = u + 0x7FFFu + ((u >> 16) & 1u);
        out[i] = (unsigned short)(r >> 16);
    }
}

// ---------------------------------------------------------------------------
// 2) Transpose W_dec [768,16384] -> W_decT [16384,768] (tiled via LDS)
// ---------------------------------------------------------------------------
__global__ void sae_transpose_wdec(const float* __restrict__ in,
                                   float* __restrict__ out) {
    __shared__ float tile[32][33];
    const int l0 = blockIdx.x * 32;   // along 16384
    const int d0 = blockIdx.y * 32;   // along 768
    const int tx = threadIdx.x;       // 0..31
    const int ty = threadIdx.y;       // 0..7
#pragma unroll
    for (int i = 0; i < 32; i += 8) {
        tile[ty + i][tx] = in[(size_t)(d0 + ty + i) * N_LATENTS + l0 + tx];
    }
    __syncthreads();
#pragma unroll
    for (int i = 0; i < 32; i += 8) {
        out[(size_t)(l0 + ty + i) * D_MODEL + d0 + tx] = tile[tx][ty + i];
    }
}

// ---------------------------------------------------------------------------
// 3) Encoder GEMM: pre[8192,16384] = x(bf16) @ W_enc(bf16)^T + b_enc
//    WG tile 128(M) x 128(N), 8 waves as 4(M) x 2(N); each wave: 2x4 16x16
//    accumulators. K staged in double-buffered LDS tiles of 64 via the CDNA5
//    async-copy engine (global_load_async_to_lds_b128 / ASYNCcnt), padded
//    row stride of 72 halves to spread LDS banks.
// ---------------------------------------------------------------------------
#define LDS_STRIDE 72   // halves per row (64 data + 8 pad), 144B, 16B aligned
#define TILE_HALVES (128 * LDS_STRIDE)

// One 16B chunk: global -> LDS, no VGPR staging, tracked by ASYNCcnt.
// The LDS operand is the 32-bit LDS byte address; for flat pointers into the
// LDS aperture, addr[31:0] IS the LDS offset (ISA 10.2), so truncation works.
__device__ __forceinline__ void async_copy16(const unsigned short* gsrc,
                                             unsigned short* ldst) {
    unsigned int lds_off = (unsigned int)(uintptr_t)ldst;
    asm volatile("global_load_async_to_lds_b128 %0, %1, off"
                 :: "v"(lds_off), "v"(gsrc)
                 : "memory");
}

__device__ __forceinline__ void wait_async_all() {
    asm volatile("s_wait_asynccnt 0x0" ::: "memory");
}

__global__ __launch_bounds__(256) void sae_gemm_enc(
    const unsigned short* __restrict__ xb,   // [8192,768] bf16 bits
    const unsigned short* __restrict__ wb,   // [16384,768] bf16 bits
    const float* __restrict__ b_enc,         // [16384]
    float* __restrict__ pre)                 // [8192,16384]
{
    __shared__ __align__(16) unsigned short As[2][TILE_HALVES];
    __shared__ __align__(16) unsigned short Bs[2][TILE_HALVES];

    const int tid  = threadIdx.x;
    const int lane = tid & 31;
    const int w    = tid >> 5;
    const int wm   = w >> 1;    // 0..3 -> M offset wm*32
    const int wn   = w & 1;     // 0..1 -> N offset wn*64
    const int m0   = blockIdx.y * 128;
    const int n0   = blockIdx.x * 128;

    v8f acc[2][4];
#pragma unroll
    for (int s = 0; s < 2; ++s)
#pragma unroll
        for (int t = 0; t < 4; ++t)
            acc[s][t] = (v8f){0.f, 0.f, 0.f, 0.f, 0.f, 0.f, 0.f, 0.f};

    const int r    = lane & 15;   // row/col within 16-subtile
    const int hsel = lane >> 4;   // half-select per ISA wave32 layouts

    // Per-thread staging geometry: 1024 16B chunks per matrix tile, 4/thread.
    const int NK = D_MODEL / 64;  // 12 K-tiles

    // Prologue: stage K-tile 0 into buffer 0.
#pragma unroll
    for (int i = 0; i < 4; ++i) {
        const int chunk = tid + i * 256;
        const int row   = chunk >> 3;
        const int part  = chunk & 7;
        async_copy16(xb + (size_t)(m0 + row) * D_MODEL + part * 8,
                     &As[0][row * LDS_STRIDE + part * 8]);
        async_copy16(wb + (size_t)(n0 + row) * D_MODEL + part * 8,
                     &Bs[0][row * LDS_STRIDE + part * 8]);
    }
    wait_async_all();
    __syncthreads();

    for (int kt = 0; kt < NK; ++kt) {
        const int cur = kt & 1;
        // Stage next K-tile into the other buffer while computing this one.
        if (kt + 1 < NK) {
            const int k0n = (kt + 1) * 64;
#pragma unroll
            for (int i = 0; i < 4; ++i) {
                const int chunk = tid + i * 256;
                const int row   = chunk >> 3;
                const int part  = chunk & 7;
                async_copy16(xb + (size_t)(m0 + row) * D_MODEL + k0n + part * 8,
                             &As[cur ^ 1][row * LDS_STRIDE + part * 8]);
                async_copy16(wb + (size_t)(n0 + row) * D_MODEL + k0n + part * 8,
                             &Bs[cur ^ 1][row * LDS_STRIDE + part * 8]);
            }
        }

        // --- two 16x16x32 K-steps on the current tile ---
        const unsigned short* AsC = &As[cur][0];
        const unsigned short* BsC = &Bs[cur][0];
#pragma unroll
        for (int kk = 0; kk < 64; kk += 32) {
#pragma unroll
            for (int s = 0; s < 2; ++s) {
                // A fragment: 16x32 bf16. lane<16: row m, K {kk+0..7, kk+16..23}
                //             lane>=16: row m, K {kk+8..15, kk+24..31}
                AFrag a;
                const int arow  = wm * 32 + s * 16 + r;
                const int abase = arow * LDS_STRIDE + kk + hsel * 8;
                a.q[0] = *(const uint4*)&AsC[abase];
                a.q[1] = *(const uint4*)&AsC[abase + 16];
#pragma unroll
                for (int t = 0; t < 4; ++t) {
                    // B fragment: 32x16 bf16. lane<16: col n, K {kk+0..7, kk+16..23}
                    //             lane>=16: col n, K {kk+8..15, kk+24..31}
                    AFrag b;
                    const int brow  = wn * 64 + t * 16 + r;
                    const int bbase = brow * LDS_STRIDE + kk + hsel * 8;
                    b.q[0] = *(const uint4*)&BsC[bbase];
                    b.q[1] = *(const uint4*)&BsC[bbase + 16];
                    acc[s][t] = __builtin_amdgcn_wmma_f32_16x16x32_bf16(
                        false, a.v, false, b.v, (short)0, acc[s][t], false, false);
                }
            }
        }

        // Next buffer fully copied + everyone done reading current buffer.
        wait_async_all();
        __syncthreads();
    }

    // --- store C + bias. 16x16 f32 layout: VGPR rr, lane<16 -> M=rr, lane>=16 -> M=rr+8
#pragma unroll
    for (int s = 0; s < 2; ++s) {
#pragma unroll
        for (int t = 0; t < 4; ++t) {
            const int col     = n0 + wn * 64 + t * 16 + (lane & 15);
            const float bias  = b_enc[col];
            const int rowbase = m0 + wm * 32 + s * 16 + (hsel << 3);
#pragma unroll
            for (int rr = 0; rr < 8; ++rr) {
                pre[(size_t)(rowbase + rr) * N_LATENTS + col] = acc[s][t][rr] + bias;
            }
        }
    }
}

// ---------------------------------------------------------------------------
// 4) Top-32 per row via 4-pass MSD radix select on sign-flipped float keys.
//    One workgroup (256 threads) per row. Writes full latents row and the
//    (idx, relu(val)) lists for the decoder.
// ---------------------------------------------------------------------------
__device__ inline unsigned int f2key(float f) {
    unsigned int u = __float_as_uint(f);
    return (u & 0x80000000u) ? ~u : (u | 0x80000000u);
}

__global__ __launch_bounds__(256) void sae_topk(
    const float* __restrict__ pre,   // [8192,16384]
    float* __restrict__ lat,         // [8192,16384]
    float* __restrict__ tkv,         // [8192,32]
    int*   __restrict__ tki)         // [8192,32]
{
    __shared__ unsigned int hist[256];
    __shared__ unsigned int s_need, s_prefix, s_eq, s_slot;

    const int row = blockIdx.x;
    const int tid = threadIdx.x;
    const float* prerow = pre + (size_t)row * N_LATENTS;
    float* latrow = lat + (size_t)row * N_LATENTS;

    if (tid == 0) { s_need = TOPK; s_prefix = 0u; s_eq = 0u; s_slot = 0u; }
    __syncthreads();

    for (int pass = 0; pass < 4; ++pass) {
        const int shift = 24 - 8 * pass;
        unsigned int himask = 0u;
        if (pass) himask = 0xFFFFFFFFu << (shift + 8);

        for (int i = tid; i < 256; i += 256) hist[i] = 0u;
        __syncthreads();

        const unsigned int pref = s_prefix;
        for (int i = tid; i < N_LATENTS; i += 256) {
            unsigned int k = f2key(prerow[i]);
            if (pass == 0 || ((k & himask) == (pref & himask)))
                atomicAdd(&hist[(k >> shift) & 0xFFu], 1u);
        }
        __syncthreads();

        if (tid == 0) {
            unsigned int need = s_need;
            unsigned int cum = 0u;
            int j = 255;
            for (; j >= 0; --j) {
                unsigned int c = hist[j];
                if (cum + c >= need) break;
                cum += c;
            }
            if (j < 0) j = 0;  // safety
            s_prefix |= ((unsigned int)j) << shift;
            s_need = need - cum;
        }
        __syncthreads();
    }

    const unsigned int T      = s_prefix;  // key of the 32nd largest element
    const unsigned int needEq = s_need;    // #elements equal to T to include

    for (int i = tid; i < N_LATENTS; i += 256) {
        const float f = prerow[i];
        const unsigned int k = f2key(f);
        float outv = 0.f;
        bool take = false;
        if (k > T) {
            take = true;
        } else if (k == T) {
            unsigned int pos = atomicAdd(&s_eq, 1u);
            if (pos < needEq) take = true;   // ties at exact threshold: measure-zero
        }
        if (take) {
            outv = fmaxf(f, 0.f);
            unsigned int slot = atomicAdd(&s_slot, 1u);
            tki[(size_t)row * TOPK + slot] = i;
            tkv[(size_t)row * TOPK + slot] = outv;
        }
        latrow[i] = outv;
    }
}

// ---------------------------------------------------------------------------
// 5) Sparse decoder: rec[row, :] = b_dec + sum_j val_j * W_decT[idx_j, :]
//    Pairs sorted by index first so FP accumulation order is deterministic.
// ---------------------------------------------------------------------------
__global__ __launch_bounds__(256) void sae_decode(
    const float* __restrict__ tkv,
    const int*   __restrict__ tki,
    const float* __restrict__ wdT,    // [16384,768]
    const float* __restrict__ b_dec,  // [768]
    float* __restrict__ rec)          // [8192,768]
{
    __shared__ float vv[TOPK];
    __shared__ int   id[TOPK];
    const int row = blockIdx.x;
    const int tid = threadIdx.x;

    if (tid < TOPK) {
        vv[tid] = tkv[(size_t)row * TOPK + tid];
        id[tid] = tki[(size_t)row * TOPK + tid];
    }
    __syncthreads();

    if (tid == 0) {  // insertion sort 32 pairs by index -> deterministic order
        for (int a = 1; a < TOPK; ++a) {
            int ia = id[a]; float va = vv[a]; int b = a - 1;
            while (b >= 0 && id[b] > ia) { id[b+1] = id[b]; vv[b+1] = vv[b]; --b; }
            id[b+1] = ia; vv[b+1] = va;
        }
    }
    __syncthreads();

    for (int d = tid; d < D_MODEL; d += 256) {
        float acc = b_dec[d];
#pragma unroll 8
        for (int j = 0; j < TOPK; ++j)
            acc += vv[j] * wdT[(size_t)id[j] * D_MODEL + d];
        rec[(size_t)row * D_MODEL + d] = acc;
    }
}

// ---------------------------------------------------------------------------
// Launcher
// ---------------------------------------------------------------------------
extern "C" void kernel_launch(void* const* d_in, const int* in_sizes, int n_in,
                              void* d_out, int out_size, void* d_ws, size_t ws_size,
                              hipStream_t stream) {
    (void)in_sizes; (void)n_in; (void)out_size; (void)ws_size;

    const float* x     = (const float*)d_in[0];   // [8192,768]
    const float* W_enc = (const float*)d_in[1];   // [16384,768]
    const float* b_enc = (const float*)d_in[2];   // [16384]
    const float* W_dec = (const float*)d_in[3];   // [768,16384]
    const float* b_dec = (const float*)d_in[4];   // [768]

    // Workspace layout (bytes), all 16B aligned:
    //   xb   : 8192*768*2   = 12,582,912
    //   wb   : 16384*768*2  = 25,165,824
    //   wdT  : 16384*768*4  = 50,331,648
    //   tkv  : 8192*32*4    =  1,048,576
    //   tki  : 8192*32*4    =  1,048,576   (total ~90.2 MB)
    char* ws = (char*)d_ws;
    unsigned short* xb  = (unsigned short*)(ws);
    unsigned short* wb  = (unsigned short*)(ws + 12582912);
    float*          wdT = (float*)(ws + 37748736);
    float*          tkv = (float*)(ws + 88080384);
    int*            tki = (int*)(ws + 89128960);

    // d_out: latents [8192*16384] | reconstructed [8192*768] | pre [8192*16384]
    float* out = (float*)d_out;
    float* lat = out;
    float* rec = out + (size_t)N_ROWS * N_LATENTS;
    float* pre = rec + (size_t)N_ROWS * D_MODEL;

    // 1) converts
    {
        int n = N_ROWS * D_MODEL;       // 6,291,456
        sae_f32_to_bf16<<<(n + 255) / 256, 256, 0, stream>>>(x, xb, n);
    }
    {
        int n = N_LATENTS * D_MODEL;    // 12,582,912
        sae_f32_to_bf16<<<(n + 255) / 256, 256, 0, stream>>>(W_enc, wb, n);
    }

    // 2) W_dec transpose
    {
        dim3 grid(N_LATENTS / 32, D_MODEL / 32);  // (512, 24)
        dim3 block(32, 8);
        sae_transpose_wdec<<<grid, block, 0, stream>>>(W_dec, wdT);
    }

    // 3) encoder GEMM (bf16 WMMA, f32 accumulate, async-LDS double buffering)
    {
        dim3 grid(N_LATENTS / 128, N_ROWS / 128);  // (128, 64)
        sae_gemm_enc<<<grid, 256, 0, stream>>>(xb, wb, b_enc, pre);
    }

    // 4) top-32 select + latents scatter
    sae_topk<<<N_ROWS, 256, 0, stream>>>(pre, lat, tkv, tki);

    // 5) sparse decoder
    sae_decode<<<N_ROWS, 256, 0, stream>>>(tkv, tki, wdT, b_dec, rec);
}
